// Faster_RCNN_33560874451189
// MI455X (gfx1250) — compile-verified, matrix operand order checked by reading, not run on hardware
//
#include <hip/hip_runtime.h>
#include <math.h>

// ---------------------------------------------------------------------------
// Types for CDNA5 WMMA (wave32, 16x16x32 bf16 -> f32)
// ---------------------------------------------------------------------------
typedef __bf16 bf16_t;
typedef __attribute__((ext_vector_type(16))) __bf16 v16bf;
typedef __attribute__((ext_vector_type(8)))  __bf16 v8bf;
typedef __attribute__((ext_vector_type(8)))  float  v8f;

#define CT 256   // threads per block (8 wave32)
#define MT 256   // M tile (spatial positions) == CT: one row per thread
#define NT 64    // N tile (output channels)
#define KT 32    // K step (bf16 WMMA K)
#define KP (KT + 8)   // padded LDS row: 40 el = 80B, 16B-aligned rows

#define USE_ASYNC_B 1   // stage full B tiles with global_load_async_to_lds_b128

static __device__ __forceinline__ unsigned short f2bf(float f)
{
    union { __bf16 b; unsigned short u; } cv;
    cv.b = (__bf16)f;
    return cv.u;
}

// ---------------------------------------------------------------------------
// Weight pre-conversion + transpose: w[k][n] f32 -> o[n][k] bf16
// (once per layer, into ws scratch; makes B tiles byte-copyable along K)
// ---------------------------------------------------------------------------
__global__ void w2bf_t_kernel(const float* __restrict__ w, unsigned short* __restrict__ o,
                              int K, int Cout)
{
    long long i = (long long)blockIdx.x * blockDim.x + threadIdx.x;
    long long n = (long long)K * Cout;
    if (i >= n) return;
    int nn = (int)(i / K);
    int kk = (int)(i - (long long)nn * K);
    o[i] = f2bf(w[(long long)kk * Cout + nn]);
}

// ---------------------------------------------------------------------------
// Implicit-GEMM conv (SAME, ksize 1 or 3) / plain GEMM via H=M, W=1, ks=1.
//   A[m][k] = in[(y+ky-pad)*W + (x+kx-pad)][cin],  k = (ky*ks+kx)*Cin + cin
//   B[k][n] = wbt[n*K + k]   (bf16, pre-transposed)
//   out[m*Cout + n] = relu(A@B + bias[n])
// Block tile 256x64; each wave: 2 x (16M x 64N) -> 8 WMMA / K-step.
// All fragments load as 2 contiguous 16B LDS reads (ds_load_b128).
// ---------------------------------------------------------------------------
__global__ __launch_bounds__(CT)
void conv_wmma_kernel(const float* __restrict__ in, const unsigned short* __restrict__ wbt,
                      const float* __restrict__ bias, float* __restrict__ out,
                      int H, int W, int Cin, int Cout, int ks, int relu)
{
    __shared__ bf16_t As[MT][KP];    // A tile, row-major (m, k)
    __shared__ bf16_t Bst[NT][KP];   // B tile, TRANSPOSED (n, k)

    const int tid  = threadIdx.x;
    const int lane = tid & 31;
    const int wave = tid >> 5;
    const int M    = H * W;
    const int K    = ks * ks * Cin;
    const int pad  = ks >> 1;
    const int m0   = blockIdx.x * MT;
    const int n0   = blockIdx.y * NT;

    const int mrow = lane & 15;   // M row (A,C,D) / N col (B,C,D) within 16
    const int half = lane >> 4;   // upper/lower K-octet select

    // this thread's output row (A staging): one division for the whole kernel
    const int  mg  = m0 + tid;
    const bool mok = (mg < M);
    int y = 0, x = 0;
    if (mok) { y = mg / W; x = mg - y * W; }

    // incremental decomposition of k0 -> (cin, kx, ky); uniform across threads
    int cc0 = 0, kx0 = 0, ky0 = 0;

    v8f acc[2][4];
    v8f zero = {0.f, 0.f, 0.f, 0.f, 0.f, 0.f, 0.f, 0.f};
#pragma unroll
    for (int t = 0; t < 2; ++t)
#pragma unroll
        for (int j = 0; j < 4; ++j) acc[t][j] = zero;

    for (int k0 = 0; k0 < K; k0 += KT) {
        // ---- stage A row (1 x KT per thread), four 16B LDS stores ---------
        {
            unsigned pk[KT / 2];
            int c2 = cc0, kx2 = kx0, ky2 = ky0;
#pragma unroll
            for (int ii = 0; ii < KT / 2; ++ii) {
                unsigned pack = 0;
#pragma unroll
                for (int h = 0; h < 2; ++h) {
                    int k = k0 + 2 * ii + h;
                    float v = 0.0f;
                    if (mok && k < K) {
                        int yy = y + ky2 - pad;
                        int xx = x + kx2 - pad;
                        if (yy >= 0 && yy < H && xx >= 0 && xx < W)
                            v = in[(long long)(yy * W + xx) * Cin + c2];
                    }
                    pack |= ((unsigned)f2bf(v)) << (16 * h);
                    if (++c2 == Cin) { c2 = 0; if (++kx2 == ks) { kx2 = 0; ++ky2; } }
                }
                pk[ii] = pack;
            }
            cc0 = c2; kx0 = kx2; ky0 = ky2;   // advanced by exactly KT
            uint4* arow = (uint4*)&As[tid][0];
#pragma unroll
            for (int q = 0; q < KT / 8; ++q)
                arow[q] = make_uint4(pk[4 * q], pk[4 * q + 1], pk[4 * q + 2], pk[4 * q + 3]);
        }

        // ---- stage B tile (NT x KT bf16, transposed) ----------------------
        const bool fullB = (k0 + KT <= K) && (n0 + NT <= Cout);
#if USE_ASYNC_B
        if (fullB) {
            // 256 threads x 16B = whole 64x32 bf16 tile, via async DMA to LDS
            int n  = tid >> 2;            // 0..63
            int kc = (tid & 3) * 8;       // 8 bf16 = 16B chunk
            unsigned lds          = (unsigned)(size_t)&Bst[n][kc];
            unsigned long long gb = (unsigned long long)(size_t)wbt;
            unsigned goff = (unsigned)(((long long)(n0 + n) * K + k0 + kc) * 2);
            asm volatile("global_load_async_to_lds_b128 %0, %1, %2"
                         :: "v"(lds), "v"(goff), "s"(gb) : "memory");
        } else
#endif
        {
#pragma unroll
            for (int i = 0; i < (KT * NT) / CT; ++i) {
                int idx = i * CT + tid;
                int n   = idx >> 5;          // KT == 32
                int kk  = idx & (KT - 1);
                int k   = k0 + kk;
                int ng  = n0 + n;
                unsigned short v = 0;
                if (k < K && ng < Cout)
                    v = wbt[(long long)ng * K + k];
                *(unsigned short*)&Bst[n][kk] = v;
            }
        }
        // prefetch next K-step weight bytes into GL2 (global_prefetch_b8)
        if (k0 + KT < K)
            __builtin_prefetch(&wbt[(long long)n0 * K + k0 + KT], 0, 1);
#if USE_ASYNC_B
        if (fullB)
            asm volatile("s_wait_asynccnt 0x0" ::: "memory");
#endif
        __syncthreads();

        // ---- A fragments: two contiguous 16B runs per lane (ISA 7.12.2) ---
        const int amBase = wave * 16 + mrow;
        v16bf afrag[2];
#pragma unroll
        for (int t = 0; t < 2; ++t) {
            const int am = t * 128 + amBase;
            v8bf lo = *(const v8bf*)&As[am][half * 8];
            v8bf hi = *(const v8bf*)&As[am][16 + half * 8];
            afrag[t] = __builtin_shufflevector(lo, hi, 0, 1, 2, 3, 4, 5, 6, 7,
                                               8, 9, 10, 11, 12, 13, 14, 15);
        }
        // ---- 4 N sub-tiles: shared B fragment, 2 WMMAs each ---------------
#pragma unroll
        for (int j = 0; j < 4; ++j) {
            const int bn = j * 16 + mrow;
            v8bf lo = *(const v8bf*)&Bst[bn][half * 8];
            v8bf hi = *(const v8bf*)&Bst[bn][16 + half * 8];
            v16bf bfrag = __builtin_shufflevector(lo, hi, 0, 1, 2, 3, 4, 5, 6, 7,
                                                  8, 9, 10, 11, 12, 13, 14, 15);
#pragma unroll
            for (int t = 0; t < 2; ++t)
                acc[t][j] = __builtin_amdgcn_wmma_f32_16x16x32_bf16(
                    false, afrag[t], false, bfrag, (short)0, acc[t][j], false, false);
        }
        __syncthreads();
    }

    // ---- epilogue: bias + ReLU, C/D layout m = r + 8*half, n = lane&15 ----
#pragma unroll
    for (int t = 0; t < 2; ++t) {
#pragma unroll
        for (int j = 0; j < 4; ++j) {
            int ng = n0 + j * 16 + mrow;
            if (ng >= Cout) continue;
            float bv = bias[ng];
#pragma unroll
            for (int r = 0; r < 8; ++r) {
                int mgo = m0 + t * 128 + wave * 16 + half * 8 + r;
                if (mgo < M) {
                    float v = acc[t][j][r] + bv;
                    if (relu) v = fmaxf(v, 0.0f);
                    out[(long long)mgo * Cout + ng] = v;
                }
            }
        }
    }
}

// ---------------------------------------------------------------------------
// 2x2 max-pool, stride 2, NHWC (H,W even)
// ---------------------------------------------------------------------------
__global__ void maxpool_kernel(const float* __restrict__ in, float* __restrict__ out,
                               int H, int W, int C)
{
    int Ho = H >> 1, Wo = W >> 1;
    long long idx = (long long)blockIdx.x * blockDim.x + threadIdx.x;
    long long total = (long long)Ho * Wo * C;
    if (idx >= total) return;
    int c = (int)(idx % C);
    long long t = idx / C;
    int x = (int)(t % Wo);
    int y = (int)(t / Wo);
    float m = -INFINITY;
#pragma unroll
    for (int dy = 0; dy < 2; ++dy)
#pragma unroll
        for (int dx = 0; dx < 2; ++dx)
            m = fmaxf(m, in[((long long)(2 * y + dy) * W + (2 * x + dx)) * C + c]);
    out[((long long)y * Wo + x) * C + c] = m;
}

// ---------------------------------------------------------------------------
// Proposals: softmax over 18 logits, anchor synthesis, box decode + clip.
// One thread per anchor (9216 = 32*32*9). Output boxes are (y1,x1,y2,x2).
// ---------------------------------------------------------------------------
__global__ void proposal_kernel(const float* __restrict__ cls,   // 1024 x 18
                                const float* __restrict__ box,   // 1024 x 36
                                float* __restrict__ props,       // 9216 x 4 (yx)
                                float* __restrict__ scores)      // 9216
{
    int i = blockIdx.x * blockDim.x + threadIdx.x;
    if (i >= 9216) return;
    int loc = i / 9, a = i - loc * 9;
    int y = loc >> 5, x = loc & 31;

    const float* s = cls + loc * 18;
    float mx = s[0];
#pragma unroll
    for (int k = 1; k < 18; ++k) mx = fmaxf(mx, s[k]);
    float sum = 0.f;
#pragma unroll
    for (int k = 0; k < 18; ++k) sum += __expf(s[k] - mx);
    scores[i] = __expf(s[9 + a] - mx) / sum;

    // anchors: base 16, ratios (1,0.5,2) major, scales (1,3,5) minor
    int rr = a / 3, sc = a - rr * 3;
    float ratio = (rr == 0) ? 1.0f : (rr == 1 ? 0.5f : 2.0f);
    float scale = (sc == 0) ? 1.0f : (sc == 1 ? 3.0f : 5.0f);
    float wsr = roundf(sqrtf(256.0f / ratio));
    float hsr = roundf(wsr * ratio);
    float W2 = wsr * scale, H2 = hsr * scale;
    float shx = (float)x * 16.0f, shy = (float)y * 16.0f;
    float ax1 = 7.5f - 0.5f * (W2 - 1.0f) + shx;
    float ay1 = 7.5f - 0.5f * (H2 - 1.0f) + shy;
    float ax2 = 7.5f + 0.5f * (W2 - 1.0f) + shx;
    float ay2 = 7.5f + 0.5f * (H2 - 1.0f) + shy;

    float w  = ax2 - ax1 + 1.0f;
    float h  = ay2 - ay1 + 1.0f;
    float cx = ax1 + 0.5f * w;
    float cy = ay1 + 0.5f * h;
    const float* d = box + i * 4;
    float pcx = d[0] * w + cx;
    float pcy = d[1] * h + cy;
    float pw  = __expf(d[2]) * w;
    float ph  = __expf(d[3]) * h;
    float px1 = fminf(fmaxf(pcx - 0.5f * pw, 0.0f), 511.0f);
    float py1 = fminf(fmaxf(pcy - 0.5f * ph, 0.0f), 511.0f);
    float px2 = fminf(fmaxf(pcx + 0.5f * pw, 0.0f), 511.0f);
    float py2 = fminf(fmaxf(pcy + 0.5f * ph, 0.0f), 511.0f);
    props[i * 4 + 0] = py1;
    props[i * 4 + 1] = px1;
    props[i * 4 + 2] = py2;
    props[i * 4 + 3] = px2;
}

// ---------------------------------------------------------------------------
// Fixed-iteration NMS: 400 sequential argmax + suppression rounds.
// Single 256-thread block; argmax ties resolved to lowest index (jnp.argmax).
// ---------------------------------------------------------------------------
#define NMS_N 9216
__global__ void nms_kernel(const float* __restrict__ boxes, const float* __restrict__ scores,
                           float* __restrict__ sw, int* __restrict__ idxs,
                           float* __restrict__ validv)
{
    __shared__ float sVal[256];
    __shared__ int   sIdx[256];
    __shared__ int   sJ;
    int tid = threadIdx.x;
    for (int i = tid; i < NMS_N; i += 256) sw[i] = scores[i];
    __syncthreads();

    for (int it = 0; it < 400; ++it) {
        float bv = -3.0e38f;
        int   bi = 0;
        for (int i = tid; i < NMS_N; i += 256) {
            float v = sw[i];
            if (v > bv) { bv = v; bi = i; }
        }
        sVal[tid] = bv; sIdx[tid] = bi;
        __syncthreads();
        for (int s = 128; s > 0; s >>= 1) {
            if (tid < s) {
                float ov = sVal[tid + s]; int oi = sIdx[tid + s];
                if (ov > sVal[tid] || (ov == sVal[tid] && oi < sIdx[tid])) {
                    sVal[tid] = ov; sIdx[tid] = oi;
                }
            }
            __syncthreads();
        }
        if (tid == 0) {
            sJ = sIdx[0];
            idxs[it]   = sIdx[0];
            validv[it] = (sVal[0] > -5.0e8f) ? 1.0f : 0.0f;
        }
        __syncthreads();
        int j = sJ;
        float jy1 = boxes[j * 4 + 0], jx1 = boxes[j * 4 + 1];
        float jy2 = boxes[j * 4 + 2], jx2 = boxes[j * 4 + 3];
        float ja  = (jy2 - jy1) * (jx2 - jx1);
        for (int i = tid; i < NMS_N; i += 256) {
            float y1 = boxes[i * 4 + 0], x1 = boxes[i * 4 + 1];
            float y2 = boxes[i * 4 + 2], x2 = boxes[i * 4 + 3];
            float yy1 = fmaxf(jy1, y1), xx1 = fmaxf(jx1, x1);
            float yy2 = fminf(jy2, y2), xx2 = fminf(jx2, x2);
            float inter = fmaxf(yy2 - yy1, 0.0f) * fmaxf(xx2 - xx1, 0.0f);
            float iou = inter / (ja + (y2 - y1) * (x2 - x1) - inter + 1e-8f);
            float v = sw[i];
            if (iou > 0.5f) v = -1.0e9f;
            if (i == j)     v = -1.0e9f;
            sw[i] = v;
        }
        __syncthreads();
    }
}

// ---------------------------------------------------------------------------
// Gather selected boxes: rois (padded), rpn_scores, normalized boxes for crop.
// ---------------------------------------------------------------------------
__global__ void select_kernel(const float* __restrict__ props, const float* __restrict__ scores,
                              const int* __restrict__ idxs, const float* __restrict__ validv,
                              float* __restrict__ rois, float* __restrict__ rscore,
                              float* __restrict__ normb)
{
    int i = blockIdx.x * blockDim.x + threadIdx.x;
    if (i >= 400) return;
    int j = idxs[i];
    float vm = validv[i];
    float b0 = props[j * 4 + 0] * vm;
    float b1 = props[j * 4 + 1] * vm;
    float b2 = props[j * 4 + 2] * vm;
    float b3 = props[j * 4 + 3] * vm;
    rois[i * 5 + 0] = 0.0f;
    rois[i * 5 + 1] = b0; rois[i * 5 + 2] = b1;
    rois[i * 5 + 3] = b2; rois[i * 5 + 4] = b3;
    rscore[i] = scores[j] * vm;
    normb[i * 4 + 0] = b0 * (1.0f / 512.0f);
    normb[i * 4 + 1] = b1 * (1.0f / 512.0f);
    normb[i * 4 + 2] = b2 * (1.0f / 512.0f);
    normb[i * 4 + 3] = b3 * (1.0f / 512.0f);
}

// ---------------------------------------------------------------------------
// crop_and_resize(feature 32x32x512 -> 14x14) fused with 2x2 maxpool -> 7x7.
// Writes flat[roi*25088 + (py*7+px)*512 + c].
// ---------------------------------------------------------------------------
__global__ void crop_pool_kernel(const float* __restrict__ feat,
                                 const float* __restrict__ normb,
                                 float* __restrict__ flat)
{
    long long idx = (long long)blockIdx.x * blockDim.x + threadIdx.x;
    const long long total = 400LL * 7 * 7 * 512;
    if (idx >= total) return;
    int c = (int)(idx % 512);
    long long t = idx / 512;
    int px = (int)(t % 7); t /= 7;
    int py = (int)(t % 7);
    int roi = (int)(t / 7);

    float y1 = normb[roi * 4 + 0], x1 = normb[roi * 4 + 1];
    float y2 = normb[roi * 4 + 2], x2 = normb[roi * 4 + 3];
    float dy = (y2 - y1) * (31.0f / 13.0f);
    float dx = (x2 - x1) * (31.0f / 13.0f);

    float m = -INFINITY;
#pragma unroll
    for (int sy = 0; sy < 2; ++sy) {
#pragma unroll
        for (int sx = 0; sx < 2; ++sx) {
            int iy = 2 * py + sy, ix = 2 * px + sx;
            float ys = y1 * 31.0f + (float)iy * dy;
            float xs = x1 * 31.0f + (float)ix * dx;
            float y0f = floorf(ys), x0f = floorf(xs);
            float wy = ys - y0f, wx = xs - x0f;
            int y0i = min(max((int)y0f, 0), 31);
            int y1i = min(y0i + 1, 31);
            int x0i = min(max((int)x0f, 0), 31);
            int x1i = min(x0i + 1, 31);
            float v00 = feat[(y0i * 32 + x0i) * 512 + c];
            float v01 = feat[(y0i * 32 + x1i) * 512 + c];
            float v10 = feat[(y1i * 32 + x0i) * 512 + c];
            float v11 = feat[(y1i * 32 + x1i) * 512 + c];
            float v = (1.f - wy) * (1.f - wx) * v00 + (1.f - wy) * wx * v01 +
                      wy * (1.f - wx) * v10 + wy * wx * v11;
            bool mask = (ys >= 0.f && ys <= 31.f && xs >= 0.f && xs <= 31.f);
            v = mask ? v : 0.f;
            m = fmaxf(m, v);
        }
    }
    flat[(long long)roi * 25088 + (py * 7 + px) * 512 + c] = m;
}

// ---------------------------------------------------------------------------
// Host orchestration
// ---------------------------------------------------------------------------
extern "C" void kernel_launch(void* const* d_in, const int* in_sizes, int n_in,
                              void* d_out, int out_size, void* d_ws, size_t ws_size,
                              hipStream_t stream)
{
    (void)in_sizes; (void)n_in; (void)out_size; (void)ws_size;

    const float* image = (const float*)d_in[0];
    const float* vgg_w[13];
    const float* vgg_b[13];
    for (int i = 0; i < 13; ++i) {
        vgg_w[i] = (const float*)d_in[1 + i];
        vgg_b[i] = (const float*)d_in[14 + i];
    }
    const float* rpn_conv_w = (const float*)d_in[27];
    const float* rpn_conv_b = (const float*)d_in[28];
    const float* rpn_cls_w  = (const float*)d_in[29];
    const float* rpn_cls_b  = (const float*)d_in[30];
    const float* rpn_box_w  = (const float*)d_in[31];
    const float* rpn_box_b  = (const float*)d_in[32];
    const float* cls_w      = (const float*)d_in[33];
    const float* cls_b      = (const float*)d_in[34];
    const float* box_w      = (const float*)d_in[35];
    const float* box_b      = (const float*)d_in[36];

    // workspace layout (float elements)
    float* ws      = (float*)d_ws;
    float* big0    = ws;                        // 16,777,216
    float* big1    = ws + 16777216LL;           // 16,777,216
    float* rpnfeat = ws + 33554432LL;           // 524,288
    float* props   = ws + 34078720LL;           // 36,864
    float* scores  = props + 36864;             // 9,216
    float* swbuf   = scores + 9216;             // 9,216
    int*   idxs    = (int*)(swbuf + 9216);      // 400
    float* validv  = (float*)(idxs + 400);      // 400
    float* normb   = validv + 400;              // 1,600
    unsigned short* wbf = (unsigned short*)(normb + 1600); // 2,359,296 bf16 max

    // output layout (concatenated flat, reference return order)
    float* out          = (float*)d_out;
    float* out_rpn_box  = out;          // 32*32*36 = 36864
    float* out_rpn_cls  = out + 36864;  // 32*32*18 = 18432
    float* out_bbox     = out + 55296;  // 400*84   = 33600
    float* out_cls      = out + 88896;  // 400*21   = 8400
    float* out_rois     = out + 97296;  // 400*5    = 2000
    float* out_rscore   = out + 99296;  // 400

    auto conv = [&](const float* in, const float* w, const float* b, float* o,
                    int H, int W, int Cin, int Cout, int ks, int relu) {
        int K = ks * ks * Cin;
        long long nk = (long long)K * Cout;
        w2bf_t_kernel<<<(int)((nk + 255) / 256), 256, 0, stream>>>(w, wbf, K, Cout);
        int M = H * W;
        dim3 grid((M + MT - 1) / MT, (Cout + NT - 1) / NT);
        conv_wmma_kernel<<<grid, CT, 0, stream>>>(in, wbf, b, o, H, W, Cin, Cout, ks, relu);
    };

    // ---- VGG-16 backbone --------------------------------------------------
    static const int cfg[17] = {64, 64, 0, 128, 128, 0, 256, 256, 256, 0,
                                512, 512, 512, 0, 512, 512, 512};
    const float* cur = image;
    int H = 512, W = 512, C = 3, wi = 0;
    for (int s = 0; s < 17; ++s) {
        float* nxt = (cur == big0) ? big1 : big0;
        if (cfg[s] == 0) {
            int total = (H / 2) * (W / 2) * C;
            maxpool_kernel<<<(total + 255) / 256, 256, 0, stream>>>(cur, nxt, H, W, C);
            H /= 2; W /= 2;
        } else {
            conv(cur, vgg_w[wi], vgg_b[wi], nxt, H, W, C, cfg[s], 3, 1);
            C = cfg[s]; ++wi;
        }
        cur = nxt;
    }
    const float* feature = cur;                       // 32x32x512
    float* flatbuf = (cur == big0) ? big1 : big0;     // reuse idle ping buffer

    // ---- RPN --------------------------------------------------------------
    conv(feature, rpn_conv_w, rpn_conv_b, rpnfeat, 32, 32, 512, 512, 3, 1);
    conv(rpnfeat, rpn_cls_w, rpn_cls_b, out_rpn_cls, 32, 32, 512, 18, 1, 0);
    conv(rpnfeat, rpn_box_w, rpn_box_b, out_rpn_box, 32, 32, 512, 36, 1, 0);

    // ---- proposals / NMS / ROI select ------------------------------------
    proposal_kernel<<<36, 256, 0, stream>>>(out_rpn_cls, out_rpn_box, props, scores);
    nms_kernel<<<1, 256, 0, stream>>>(props, scores, swbuf, idxs, validv);
    select_kernel<<<2, 256, 0, stream>>>(props, scores, idxs, validv,
                                         out_rois, out_rscore, normb);

    // ---- crop_and_resize + 2x2 maxpool -> flat (400 x 25088) -------------
    {
        long long total = 400LL * 7 * 7 * 512;
        crop_pool_kernel<<<(int)((total + 255) / 256), 256, 0, stream>>>(feature, normb, flatbuf);
    }

    // ---- FC heads as WMMA GEMMs (M=400, K=25088) -------------------------
    conv(flatbuf, cls_w, cls_b, out_cls,  400, 1, 25088, 21, 1, 0);
    conv(flatbuf, box_w, box_b, out_bbox, 400, 1, 25088, 84, 1, 0);
}
